// RadixAttention_74612171866258
// MI455X (gfx1250) — compile-verified, matrix operand order checked by reading, not run on hardware
//
#include <hip/hip_runtime.h>

#define B_ 4
#define E_ 512
#define P_ 1536
#define H_ 32
#define HK_ 8
#define G_ 4
#define D_ 128
#define QSCALE 0.08838834764831845f   // 1/sqrt(128)
#define NPREF (P_ / 32)               // 48 prefix key-blocks
#define VTD (D_ + 16)                 // V^T rows incl. ones-column tile

typedef __attribute__((ext_vector_type(16))) __bf16 v16bf;
typedef __attribute__((ext_vector_type(8)))  float  v8f;
typedef __attribute__((ext_vector_type(8)))  unsigned short us8;

union Frag16 {
  us8 h[2];
  v16bf v;
  unsigned short s[16];
};

// Hardware f32 -> bf16
__device__ __forceinline__ unsigned short f2bf(float f) {
  union { __bf16 b; unsigned short u; } c;
  c.b = (__bf16)f;
  return c.u;
}

// One DPP-fused max step (ctrl must be an ICE -> template param)
template <int CTRL>
__device__ __forceinline__ float dpp_max_step(float x) {
  union { float f; int i; } s, d;
  s.f = x;
  d.i = __builtin_amdgcn_update_dpp(s.i, s.i, CTRL, 0xF, 0xF, false);
  return fmaxf(x, d.f);
}
// Max across each 16-lane half, result in all lanes. Pure VALU (no DS pipe).
__device__ __forceinline__ float max_red16(float x) {
  x = dpp_max_step<0xB1>(x);    // quad_perm [1,0,3,2]  (xor 1)
  x = dpp_max_step<0x4E>(x);    // quad_perm [2,3,0,1]  (xor 2)
  x = dpp_max_step<0x141>(x);   // row_half_mirror      (pairs 4-groups)
  x = dpp_max_step<0x140>(x);   // row_mirror           (pairs 8-groups)
  return x;
}

// ---- software-pipeline staging: next tile's f32 data in registers ----
struct Stage {
  float4 kf[8];
  float4 vf[8];
};

__device__ __forceinline__ void stage_load(Stage& st, const float* __restrict__ kbase,
                                           const float* __restrict__ vbase, int tid) {
  const int r = tid >> 2;          // key row 0..31
  const int cb = (tid & 3) * 32;   // dim column base
  const float* kp = kbase + (size_t)r * (HK_ * D_) + cb;
  const float* vp = vbase + (size_t)r * (HK_ * D_) + cb;
#pragma unroll
  for (int i = 0; i < 8; ++i) {
    st.kf[i] = *(const float4*)(kp + i * 4);
    st.vf[i] = *(const float4*)(vp + i * 4);
  }
}

__device__ __forceinline__ void stage_store(const Stage& st,
                                            unsigned short* __restrict__ K_lds,
                                            unsigned short* __restrict__ Vt_lds,
                                            int tid) {
  const int r = tid >> 2;
  const int cb = (tid & 3) * 32;
#pragma unroll
  for (int i = 0; i < 8; ++i) {
    union { unsigned short s[4]; uint2 u; } pk;
    pk.s[0] = f2bf(st.kf[i].x); pk.s[1] = f2bf(st.kf[i].y);
    pk.s[2] = f2bf(st.kf[i].z); pk.s[3] = f2bf(st.kf[i].w);
    *(uint2*)&K_lds[r * 128 + cb + i * 4] = pk.u;          // K row-major
    Vt_lds[(cb + i * 4 + 0) * 32 + r] = f2bf(st.vf[i].x);  // V transposed
    Vt_lds[(cb + i * 4 + 1) * 32 + r] = f2bf(st.vf[i].y);
    Vt_lds[(cb + i * 4 + 2) * 32 + r] = f2bf(st.vf[i].z);
    Vt_lds[(cb + i * 4 + 3) * 32 + r] = f2bf(st.vf[i].w);
  }
}

// K/V base pointer for pipeline block i (prefix cache first, then new tokens).
__device__ __forceinline__ const float* kv_base(const float* __restrict__ newbuf,
                                                const float* __restrict__ cachebuf,
                                                int i, int b, int hk) {
  if (i < NPREF)
    return cachebuf + ((size_t)(b * P_ + i * 32) * HK_ + hk) * D_;
  return newbuf + (size_t)(b * E_ + (i - NPREF) * 32) * (HK_ * D_) + hk * D_;
}

// One 32-key block for one wave: S = Q K^T, online softmax, O += P [V | 1].
__device__ __forceinline__ void attn_block(
    const unsigned short* __restrict__ K_lds,
    const unsigned short* __restrict__ Vt_lds,
    unsigned short* __restrict__ Pw,
    const Frag16* __restrict__ qa, v8f* __restrict__ o,  // o[0..7]=O, o[8]=rowsum
    float* __restrict__ rmax,
    int lane, int kt0, int qt0, bool mask) {
  const int l16 = lane & 15;
  const int lhalf = lane >> 4;

  // ---- S = Q K^T : batch the 4 B-frag loads, then chain 4 WMMAs ----
  v8f s[2];
#pragma unroll
  for (int ks = 0; ks < 2; ++ks) {
    const int key = ks * 16 + l16;
    Frag16 kb[4];
#pragma unroll
    for (int dc = 0; dc < 4; ++dc) {
      const int d0 = dc * 32 + lhalf * 8;
      kb[dc].h[0] = *(const us8*)&K_lds[key * 128 + d0];
      kb[dc].h[1] = *(const us8*)&K_lds[key * 128 + d0 + 16];
    }
    v8f acc;
#pragma unroll
    for (int j = 0; j < 8; ++j) acc[j] = 0.0f;
#pragma unroll
    for (int dc = 0; dc < 4; ++dc)
      acc = __builtin_amdgcn_wmma_f32_16x16x32_bf16(
          false, qa[dc].v, false, kb[dc].v, (short)0, acc, false, false);
    s[ks] = acc;
  }

  // ---- causal mask (arithmetic, uniform control flow) ----
  if (mask) {
#pragma unroll
    for (int ks = 0; ks < 2; ++ks) {
      const int kt = kt0 + ks * 16 + l16;
#pragma unroll
      for (int r = 0; r < 8; ++r) {
        const int qt = qt0 + r + 8 * lhalf;
        if (kt > qt) s[ks][r] = -1e30f;
      }
    }
  }

  // ---- online softmax (max via DPP; row sums come from the extra WMMA) ----
  float scl[8];
#pragma unroll
  for (int r = 0; r < 8; ++r) {
    const float m = max_red16(fmaxf(s[0][r], s[1][r]));
    const float nm = fmaxf(rmax[r], m);
    scl[r] = __expf(rmax[r] - nm);
    rmax[r] = nm;
    s[0][r] = __expf(s[0][r] - nm);
    s[1][r] = __expf(s[1][r] - nm);
  }

  // ---- rescale O accumulators (incl. the rowsum frag) ----
#pragma unroll
  for (int f = 0; f < 9; ++f)
#pragma unroll
    for (int r = 0; r < 8; ++r) o[f][r] *= scl[r];

  // ---- D-layout -> A-layout for P via per-wave LDS scratch ----
#pragma unroll
  for (int ks = 0; ks < 2; ++ks)
#pragma unroll
    for (int r = 0; r < 8; ++r)
      Pw[(r + 8 * lhalf) * 32 + ks * 16 + l16] = f2bf(s[ks][r]);
  Frag16 pa;
  pa.h[0] = *(const us8*)&Pw[l16 * 32 + lhalf * 8];
  pa.h[1] = *(const us8*)&Pw[l16 * 32 + lhalf * 8 + 16];

  // ---- O += P V : 8 dim-tiles + 1 ones-column tile (row sums) ----
#pragma unroll
  for (int g = 0; g < 2; ++g) {
    Frag16 vb[4];
#pragma unroll
    for (int t = 0; t < 4; ++t) {
      const int dim = (g * 4 + t) * 16 + l16;
      vb[t].h[0] = *(const us8*)&Vt_lds[dim * 32 + lhalf * 8];
      vb[t].h[1] = *(const us8*)&Vt_lds[dim * 32 + lhalf * 8 + 16];
    }
#pragma unroll
    for (int t = 0; t < 4; ++t)
      o[g * 4 + t] = __builtin_amdgcn_wmma_f32_16x16x32_bf16(
          false, pa.v, false, vb[t].v, (short)0, o[g * 4 + t], false, false);
  }
  {
    Frag16 vb;   // dims 128..143: col 0 is ones -> o[8][:,0] accumulates sum(p)
    const int dim = 128 + l16;
    vb.h[0] = *(const us8*)&Vt_lds[dim * 32 + lhalf * 8];
    vb.h[1] = *(const us8*)&Vt_lds[dim * 32 + lhalf * 8 + 16];
    o[8] = __builtin_amdgcn_wmma_f32_16x16x32_bf16(
        false, pa.v, false, vb.v, (short)0, o[8], false, false);
  }
}

__global__ __launch_bounds__(128) void radix_attn_kernel(
    const float* __restrict__ q, const float* __restrict__ k,
    const float* __restrict__ v, const float* __restrict__ kc,
    const float* __restrict__ vc, float* __restrict__ out) {
  __shared__ unsigned short K_lds[2][32 * 128];    // double-buffered K (bf16)
  __shared__ unsigned short Vt_lds[2][VTD * 32];   // double-buffered V^T + ones
  __shared__ unsigned short P_lds[4 * 16 * 32];    // per-wave P transpose scratch

  const int tid = threadIdx.x;
  const int wave = tid >> 5;
  const int lane = tid & 31;
  const int l16 = lane & 15;
  const int lhalf = lane >> 4;

  const int blk = blockIdx.x;       // b*(HK*32) + hk*32 + qi
  const int qi = blk & 31;
  const int hk = (blk >> 5) & 7;
  const int b = blk >> 8;
  const int qt0 = qi * 16;
  const int h = hk * G_ + wave;     // this wave's query head

  // ---- ones-column tiles (constant; written once, both buffers) ----
  if (tid < 64) {
    const int key = tid & 31;
    const int buf = tid >> 5;
    const unsigned short one = f2bf(1.0f);
#pragma unroll
    for (int dd = 0; dd < 16; ++dd)
      Vt_lds[buf][(128 + dd) * 32 + key] = (dd == 0) ? one : (unsigned short)0;
  }

  // ---- Q A-frags, pre-scaled by 1/sqrt(D), fp32 -> bf16 ----
  Frag16 qa[4];
  {
    const float* qrow = q + (size_t)(b * E_ + qt0 + l16) * (H_ * D_) + h * D_;
#pragma unroll
    for (int dc = 0; dc < 4; ++dc) {
      const int d0 = dc * 32 + lhalf * 8;
#pragma unroll
      for (int half = 0; half < 2; ++half)
#pragma unroll
        for (int i = 0; i < 8; ++i)
          qa[dc].s[half * 8 + i] = f2bf(qrow[d0 + half * 16 + i] * QSCALE);
    }
  }

  v8f o[9];                          // o[8] = softmax row-sum accumulator
#pragma unroll
  for (int f = 0; f < 9; ++f)
#pragma unroll
    for (int j = 0; j < 8; ++j) o[f][j] = 0.0f;
  float rmax[8];
#pragma unroll
  for (int r = 0; r < 8; ++r) rmax[r] = -1e30f;

  unsigned short* Pw = P_lds + wave * (16 * 32);

  // ---- software-pipelined main loop over all key blocks ----
  const int nblk = NPREF + ((qt0 + 16 + 31) >> 5);

  Stage st;
  stage_load(st, kv_base(k, kc, 0, b, hk), kv_base(v, vc, 0, b, hk), tid);

  for (int i = 0; i < nblk; ++i) {
    stage_store(st, K_lds[i & 1], Vt_lds[i & 1], tid);
    __syncthreads();   // buf[i&1] ready; transitively protects buffer reuse
    if (i + 1 < nblk)  // issue next tile's global loads before compute
      stage_load(st, kv_base(k, kc, i + 1, b, hk),
                 kv_base(v, vc, i + 1, b, hk), tid);
    const bool causal = (i >= NPREF);
    const int kt0 = causal ? (i - NPREF) * 32 : 0;
    attn_block(K_lds[i & 1], Vt_lds[i & 1], Pw, qa, o, rmax, lane,
               kt0, qt0, causal && (kt0 + 31 > qt0));
  }

  // ---- normalize (row sums live in column 0 of o[8]) and store ----
  float inv[8];
#pragma unroll
  for (int r = 0; r < 8; ++r)
    inv[r] = 1.0f / __shfl(o[8][r], lhalf * 16, 32);
  float* obase = out + (size_t)(b * E_ + qt0) * (H_ * D_) + h * D_;
#pragma unroll
  for (int f = 0; f < 8; ++f)
#pragma unroll
    for (int r = 0; r < 8; ++r)
      obase[(size_t)(r + 8 * lhalf) * (H_ * D_) + f * 16 + l16] =
          o[f][r] * inv[r];
}

extern "C" void kernel_launch(void* const* d_in, const int* in_sizes, int n_in,
                              void* d_out, int out_size, void* d_ws,
                              size_t ws_size, hipStream_t stream) {
  (void)in_sizes; (void)n_in; (void)out_size; (void)d_ws; (void)ws_size;
  const float* q = (const float*)d_in[0];
  const float* k = (const float*)d_in[1];
  const float* v = (const float*)d_in[2];
  const float* kc = (const float*)d_in[3];
  const float* vc = (const float*)d_in[4];
  dim3 grid(B_ * HK_ * (E_ / 16));   // 1024 blocks: (b, kv head, 16-query tile)
  dim3 block(128);                   // 4 waves = G query heads sharing KV
  hipLaunchKernelGGL(radix_attn_kernel, grid, block, 0, stream,
                     q, k, v, kc, vc, (float*)d_out);
}
